// Model_20547123544447
// MI455X (gfx1250) — matrix operand order repaired and checked by reading, not compiled, not run on hardware
//
#include <hip/hip_runtime.h>
#include <math.h>

typedef __attribute__((ext_vector_type(16))) __bf16 v16bf;
typedef __attribute__((ext_vector_type(8)))  float  v8f;

union AB { v16bf v; uint4 q[2]; };

__device__ __forceinline__ float bf2f(unsigned short s) {
    union { unsigned u; float f; } o; o.u = ((unsigned)s) << 16; return o.f;
}
__device__ __forceinline__ unsigned short f2bfbits(float f) {
    union { float f; unsigned u; } v; v.f = f;
    unsigned r = v.u + 0x7FFFu + ((v.u >> 16) & 1u);   // round-to-nearest-even
    return (unsigned short)(r >> 16);
}

// ---------------------------------------------------------------- conversions
__global__ void k_f32_to_bf16(const float* __restrict__ src,
                              unsigned short* __restrict__ dst, int n) {
    int i = blockIdx.x * 256 + threadIdx.x;
    if (i < n) dst[i] = f2bfbits(src[i]);
}

// ---------------------------------------------------------------- input projection
// xg[dir][m][g] = sum_k in[m][k] * W[dir][g][k] + bias[dir][g]
// M = T*B = 32768, N = 3H = 768, K in {128, 512}.
// Register-blocked 16x64 tile per wave: one A fragment feeds 4 WMMAs.
__global__ __launch_bounds__(256)
void k_xproj(const unsigned short* __restrict__ A, int K,
             const unsigned short* __restrict__ W,     // [2][768*K] row-major (g, k)
             const float* __restrict__ bias,           // [2][768]
             float* __restrict__ out)                  // [2][32768*768]
{
    const int M = 32768, NG = 12 /* 768/64 */, MT = 2048;
    const int wave = threadIdx.x >> 5;
    const int lane = threadIdx.x & 31;
    const int tid  = blockIdx.x * 8 + wave;            // task id over (dir, tm, ng)
    const int per  = MT * NG;
    const int dir  = tid / per;
    if (dir >= 2) return;                              // uniform per wave -> EXEC stays full
    const int t2 = tid % per;
    const int tm = t2 / NG, ng = t2 % NG;

    const unsigned short* Wd = W + (size_t)dir * 768 * K;
    float*                od = out + (size_t)dir * M * 768;

    const int lm = lane & 15, lh = lane >> 4;
    const unsigned short* arow = A  + (size_t)(tm * 16 + lm) * K;                // A row (M index)
    const unsigned short* brow = Wd + (size_t)(ng * 64 + lm) * K + lh * 16;      // B col base, K-major
    const size_t bstep = (size_t)16 * K;                                         // next 16-col group

    v8f acc0 = {}, acc1 = {}, acc2 = {}, acc3 = {};
    for (int k0 = 0; k0 < K; k0 += 32) {
        AB a, b0, b1, b2, b3;
        a.q[0]  = *(const uint4*)(arow + k0 + lh * 16);
        a.q[1]  = *(const uint4*)(arow + k0 + lh * 16 + 8);
        b0.q[0] = *(const uint4*)(brow + 0 * bstep + k0);
        b0.q[1] = *(const uint4*)(brow + 0 * bstep + k0 + 8);
        b1.q[0] = *(const uint4*)(brow + 1 * bstep + k0);
        b1.q[1] = *(const uint4*)(brow + 1 * bstep + k0 + 8);
        b2.q[0] = *(const uint4*)(brow + 2 * bstep + k0);
        b2.q[1] = *(const uint4*)(brow + 2 * bstep + k0 + 8);
        b3.q[0] = *(const uint4*)(brow + 3 * bstep + k0);
        b3.q[1] = *(const uint4*)(brow + 3 * bstep + k0 + 8);
        acc0 = __builtin_amdgcn_wmma_f32_16x16x32_bf16(false, a.v, false, b0.v,
                                                       (short)0, acc0, false, false);
        acc1 = __builtin_amdgcn_wmma_f32_16x16x32_bf16(false, a.v, false, b1.v,
                                                       (short)0, acc1, false, false);
        acc2 = __builtin_amdgcn_wmma_f32_16x16x32_bf16(false, a.v, false, b2.v,
                                                       (short)0, acc2, false, false);
        acc3 = __builtin_amdgcn_wmma_f32_16x16x32_bf16(false, a.v, false, b3.v,
                                                       (short)0, acc3, false, false);
    }

    v8f accs[4] = {acc0, acc1, acc2, acc3};
#pragma unroll
    for (int s = 0; s < 4; ++s) {
        const int col = ng * 64 + s * 16 + lm;
        const float bv = bias[dir * 768 + col];
#pragma unroll
        for (int r = 0; r < 8; ++r) {
            const int row = tm * 16 + r + 8 * lh;
            od[(size_t)row * 768 + col] = accs[s][r] + bv;
        }
    }
}

// ---------------------------------------------------------------- recurrent scan
// One block per direction. h double-buffered in LDS (bf16). Each wave computes
// r/z/n gate tiles for the same (b,j) 16x16 patch so gating is register-local.
__global__ __launch_bounds__(1024)
void k_gru_scan(const float* __restrict__ xg,            // [2][T*64*768] (incl. b_ih)
                const unsigned short* __restrict__ whh,  // [2][768*256] row-major (g, k)
                const float* __restrict__ bhh,           // [2][768]
                const float* __restrict__ h0,            // [2][64*256]
                unsigned short* __restrict__ yout,       // [T*64*512] bf16
                float* __restrict__ hfin,                // [2][64*256]
                int T)
{
    __shared__ alignas(16) unsigned short hbuf[2][64 * 256];

    const int dir = blockIdx.x;
    const bool rev = (dir == 1);
    const float*          xgd = xg  + (size_t)dir * T * 64 * 768;
    const unsigned short* whd = whh + (size_t)dir * 768 * 256;
    const float*          bhd = bhh + dir * 768;

    for (int i = threadIdx.x; i < 64 * 256; i += blockDim.x)
        hbuf[0][i] = f2bfbits(h0[dir * 64 * 256 + i]);
    __syncthreads();

    const int wave = threadIdx.x >> 5;
    const int lane = threadIdx.x & 31;
    const int lm = lane & 15, lh = lane >> 4;

    for (int step = 0; step < T; ++step) {
        const int t   = rev ? (T - 1 - step) : step;
        const int cur = step & 1, nxt = cur ^ 1;

        for (int i = 0; i < 2; ++i) {
            const int task = (wave << 1) | i;   // 0..63
            const int mt = task >> 4;           // batch tile  0..3
            const int nt = task & 15;           // hidden tile 0..15
            const int j  = nt * 16 + lm;

            // prefetch next timestep's input-gate rows while we do this step's GEMM
            if (step + 1 < T) {
                const int t2 = rev ? (t - 1) : (t + 1);
                const float* pf = xgd + ((size_t)t2 * 64 + (mt * 16 + lh * 8)) * 768 + j;
                __builtin_prefetch(pf, 0, 1);
            }

            v8f aR = {}, aZ = {}, aN = {};
            const unsigned short* hrow = &hbuf[cur][(mt * 16 + lm) * 256];
            const unsigned short* bR = whd + (size_t)(  0 + nt * 16 + lm) * 256 + lh * 16;
            const unsigned short* bZ = whd + (size_t)(256 + nt * 16 + lm) * 256 + lh * 16;
            const unsigned short* bN = whd + (size_t)(512 + nt * 16 + lm) * 256 + lh * 16;

#pragma unroll
            for (int k0 = 0; k0 < 256; k0 += 32) {
                AB a, b0, b1, b2;
                a.q[0]  = *(const uint4*)(hrow + k0 + lh * 16);
                a.q[1]  = *(const uint4*)(hrow + k0 + lh * 16 + 8);
                b0.q[0] = *(const uint4*)(bR + k0);  b0.q[1] = *(const uint4*)(bR + k0 + 8);
                b1.q[0] = *(const uint4*)(bZ + k0);  b1.q[1] = *(const uint4*)(bZ + k0 + 8);
                b2.q[0] = *(const uint4*)(bN + k0);  b2.q[1] = *(const uint4*)(bN + k0 + 8);
                aR = __builtin_amdgcn_wmma_f32_16x16x32_bf16(false, a.v, false, b0.v,
                                                             (short)0, aR, false, false);
                aZ = __builtin_amdgcn_wmma_f32_16x16x32_bf16(false, a.v, false, b1.v,
                                                             (short)0, aZ, false, false);
                aN = __builtin_amdgcn_wmma_f32_16x16x32_bf16(false, a.v, false, b2.v,
                                                             (short)0, aN, false, false);
            }

            // fused gating on the accumulators (lane owns column j, 8 rows)
            const float bhr = bhd[j], bhz = bhd[256 + j], bhn = bhd[512 + j];
#pragma unroll
            for (int r = 0; r < 8; ++r) {
                const int b = mt * 16 + r + 8 * lh;
                const float* xrow = xgd + ((size_t)t * 64 + b) * 768;
                const float xr = xrow[j], xz = xrow[256 + j], xn = xrow[512 + j];
                const float hr = aR[r] + bhr, hz = aZ[r] + bhz, hn = aN[r] + bhn;
                const float rg = 1.f / (1.f + __expf(-(xr + hr)));
                const float zg = 1.f / (1.f + __expf(-(xz + hz)));
                const float ng = tanhf(xn + rg * hn);
                const float hp = bf2f(hbuf[cur][b * 256 + j]);
                const float hv = (1.f - zg) * ng + zg * hp;
                const unsigned short hb = f2bfbits(hv);
                hbuf[nxt][b * 256 + j] = hb;
                yout[((size_t)t * 64 + b) * 512 + dir * 256 + j] = hb;
                if (step == T - 1) hfin[dir * 64 * 256 + b * 256 + j] = hv;
            }
        }
        __syncthreads();   // h[nxt] fully written before next step reads it
    }
}

// ---------------------------------------------------------------- host orchestration
extern "C" void kernel_launch(void* const* d_in, const int* in_sizes, int n_in,
                              void* d_out, int out_size, void* d_ws, size_t ws_size,
                              hipStream_t stream) {
    const float* x    = (const float*)d_in[0];   // [512,64,128]
    const float* h0   = (const float*)d_in[1];   // [12,64,256]
    const float* wih0 = (const float*)d_in[2];   // [2,768,128]
    const float* wihr = (const float*)d_in[3];   // [5,2,768,512]
    const float* whh  = (const float*)d_in[4];   // [6,2,768,256]
    const float* bih  = (const float*)d_in[5];   // [6,2,768]
    const float* bhh  = (const float*)d_in[6];   // [6,2,768]
    float* out = (float*)d_out;                  // [12,64,256]

    const int T = 512, B = 64, L = 6;

    size_t off = 0;
    auto take = [&](size_t bytes) -> char* {
        char* p = (char*)d_ws + off;
        off += (bytes + 255) & ~(size_t)255;
        return p;
    };
    unsigned short* wih0_bf = (unsigned short*)take((size_t)196608 * 2);
    unsigned short* wihr_bf = (unsigned short*)take((size_t)3932160 * 2);
    unsigned short* whh_bf  = (unsigned short*)take((size_t)2359296 * 2);
    unsigned short* x_bf    = (unsigned short*)take((size_t)4194304 * 2);
    unsigned short* ybufA   = (unsigned short*)take((size_t)T * B * 512 * 2);
    unsigned short* ybufB   = (unsigned short*)take((size_t)T * B * 512 * 2);
    float*          xg      = (float*)take((size_t)2 * T * B * 768 * 4);

    auto conv = [&](const float* s, unsigned short* d, int n) {
        k_f32_to_bf16<<<(n + 255) / 256, 256, 0, stream>>>(s, d, n);
    };
    conv(x,    x_bf,    4194304);
    conv(wih0, wih0_bf, 196608);
    conv(wihr, wihr_bf, 3932160);
    conv(whh,  whh_bf,  2359296);

    const unsigned short* in = x_bf;
    for (int layer = 0; layer < L; ++layer) {
        const int K = (layer == 0) ? 128 : 512;
        const unsigned short* wi =
            (layer == 0) ? wih0_bf : (wihr_bf + (size_t)(layer - 1) * 2 * 768 * 512);

        // 2 dirs * 2048 m-tiles * 12 n-groups / 8 waves per block = 6144 blocks
        k_xproj<<<6144, 256, 0, stream>>>(in, K, wi, bih + (size_t)layer * 2 * 768, xg);

        unsigned short* yo = (layer & 1) ? ybufB : ybufA;
        k_gru_scan<<<2, 1024, 0, stream>>>(xg,
                                           whh_bf + (size_t)layer * 2 * 768 * 256,
                                           bhh + (size_t)layer * 2 * 768,
                                           h0 + (size_t)layer * 2 * 64 * 256,
                                           yo,
                                           out + (size_t)layer * 2 * 64 * 256,
                                           T);
        in = yo;
    }
    (void)in_sizes; (void)n_in; (void)out_size; (void)ws_size;
}
